// BboxEncoder_22110491639852
// MI455X (gfx1250) — matrix operand order repaired and checked by reading, not compiled, not run on hardware
//
#include <hip/hip_runtime.h>

typedef __attribute__((ext_vector_type(16))) _Float16 v16h;
typedef __attribute__((ext_vector_type(8)))  float    v8f;

#define NWAVE   4
#define B_TOTAL 32768
#define LN_EPS  1e-5f

__device__ __forceinline__ float selu_f(float x) {
    const float a = 1.6732632423543772f, s = 1.0507009873554805f;
    return s * (x > 0.f ? x : a * (__expf(x) - 1.f));
}

union FragU { uint4 u[2]; v16h v; };
union H16U  { uint4 u[2]; _Float16 h[16]; };

// A fragment: 16x32 f16, rows mrow0..+15, cols kk..kk+31 (contiguous k: element s of lane half h is k = kk + 16h + s).
// stride = row stride in halves (multiple of 8 -> 16B aligned chunks).
__device__ __forceinline__ v16h load_a_frag(const _Float16* base, int mrow0, int kk,
                                            int lane, int stride) {
    const int row = mrow0 + (lane & 15);
    const int kb  = kk + ((lane & 16) ? 16 : 0);
    const _Float16* p = base + row * stride + kb;
    FragU t;
    t.u[0] = *reinterpret_cast<const uint4*>(p);        // K = kb .. kb+7
    t.u[1] = *reinterpret_cast<const uint4*>(p + 8);    // K = kb+8 .. kb+15
    return t.v;
}

// B fragment: 32x16 f16 from a [N][K] array; lane holds one column,
// half-waves split K 0..15 / 16..31.
__device__ __forceinline__ v16h load_b_frag(const _Float16* base, int nt, int kk,
                                            int lane, int stride) {
    const int col = nt * 16 + (lane & 15);
    const int kb  = kk + ((lane & 16) ? 16 : 0);
    const _Float16* p = base + col * stride + kb;
    FragU t;
    t.u[0] = *reinterpret_cast<const uint4*>(p);
    t.u[1] = *reinterpret_cast<const uint4*>(p + 8);
    return t.v;
}

__global__ __launch_bounds__(32 * NWAVE) void bbox_encoder_kernel(
    const float* __restrict__ x,
    const float* __restrict__ W1, const float* __restrict__ b1,
    const float* __restrict__ g1, const float* __restrict__ be1,
    const float* __restrict__ W2, const float* __restrict__ b2,
    const float* __restrict__ g2, const float* __restrict__ be2,
    const float* __restrict__ Wp, const float* __restrict__ bp,
    const float* __restrict__ gp, const float* __restrict__ bep,
    float* __restrict__ out)
{
    // f16 weight / operand staging (all uint4-accessed arrays 16B aligned,
    // padded row strides for LDS bank spread)
    __shared__ __align__(16) _Float16 sW2T[64 * 136];        // W2^T  [N][K=128]
    __shared__ __align__(16) _Float16 sWpT[64 * 136];        // Wp^T  [N][K=128]
    __shared__ __align__(16) _Float16 sW1T[64 * 40];         // W1^T  [N][K=32, zero pad k>=6]
    __shared__ __align__(16) _Float16 sE2 [NWAVE][32 * 136]; // edge feats (e1 uses cols 0..31)
    __shared__ __align__(16) _Float16 sX1h[NWAVE][16 * 72];  // x1 f16 (rows 8..15 zero)
    __shared__ __align__(16) _Float16 sFEATH[16 * 136];      // pooled feats f16 (rows 4..15 zero)
    __shared__ float sD  [NWAVE][32 * 68];                   // WMMA D tiles / Gram matrix
    __shared__ float sPD [16 * 68];                          // projection D tile
    __shared__ float sX1 [NWAVE][8 * 68];                    // conv1 out (f32)
    __shared__ float sX2 [NWAVE][8 * 68];                    // conv2 out (f32)
    __shared__ float sXP [NWAVE][32];                        // points [8][4]
    __shared__ int   sIDX[NWAVE][32];                        // knn idx [8][4]
    __shared__ float sVec[9 * 64];                           // b1,g1,be1,b2,g2,be2,bp,gp,bep

    const int tid  = threadIdx.x;
    const int lane = tid & 31;
    const int wv   = tid >> 5;
    const int box  = blockIdx.x * NWAVE + wv;

    // ---- one-time block staging ----
    for (int e = tid; e < 128 * 64; e += 32 * NWAVE) {
        int k = e >> 6, n = e & 63;
        sW2T[n * 136 + k] = (_Float16)W2[e];
        sWpT[n * 136 + k] = (_Float16)Wp[e];
    }
    for (int e = tid; e < 64 * 32; e += 32 * NWAVE) {
        int col = e >> 5, k = e & 31;
        sW1T[col * 40 + k] = (k < 6) ? (_Float16)W1[k * 64 + col] : (_Float16)0.f;
    }
    for (int e = tid; e < 16 * 136; e += 32 * NWAVE) sFEATH[e] = (_Float16)0.f;
    for (int e = tid; e < NWAVE * 16 * 72; e += 32 * NWAVE)
        (&sX1h[0][0])[e] = (_Float16)0.f;
    if (tid < 64) {
        sVec[0*64+tid] = b1[tid]; sVec[1*64+tid] = g1[tid]; sVec[2*64+tid] = be1[tid];
        sVec[3*64+tid] = b2[tid]; sVec[4*64+tid] = g2[tid]; sVec[5*64+tid] = be2[tid];
        sVec[6*64+tid] = bp[tid]; sVec[7*64+tid] = gp[tid]; sVec[8*64+tid] = bep[tid];
    }
    if (lane < 24) sXP[wv][(lane / 3) * 4 + (lane % 3)] = x[box * 24 + lane];
    __syncthreads();

    float* xp   = sXP[wv];
    int*   idxp = sIDX[wv];

    // ---- kNN for conv1 (3-dim), top-4 smallest, stable index ties ----
    if (lane < 8) {
        float d[8];
        const float xi0 = xp[lane*4+0], xi1 = xp[lane*4+1], xi2 = xp[lane*4+2];
        #pragma unroll
        for (int j = 0; j < 8; ++j) {
            float d0 = xp[j*4+0]-xi0, d1 = xp[j*4+1]-xi1, d2 = xp[j*4+2]-xi2;
            d[j] = d0*d0 + d1*d1 + d2*d2;
        }
        unsigned chosen = 0;
        #pragma unroll
        for (int s = 0; s < 4; ++s) {
            float best = 3.4e38f; int bj = 0;
            #pragma unroll
            for (int j = 0; j < 8; ++j)
                if (!((chosen >> j) & 1u) && d[j] < best) { best = d[j]; bj = j; }
            chosen |= 1u << bj;
            idxp[lane*4+s] = bj;
        }
    }
    __syncthreads();

    // ---- build e1 rows [32, K=32] f16 (6 features + zero pad) ----
    {
        const int p = lane >> 2, nn = lane & 3;
        const int j = idxp[p*4+nn];
        H16U lo, hi;
        #pragma unroll
        for (int t = 0; t < 16; ++t) { lo.h[t] = (_Float16)0.f; hi.h[t] = (_Float16)0.f; }
        lo.h[0] = (_Float16)(xp[j*4+0]-xp[p*4+0]);
        lo.h[1] = (_Float16)(xp[j*4+1]-xp[p*4+1]);
        lo.h[2] = (_Float16)(xp[j*4+2]-xp[p*4+2]);
        lo.h[3] = (_Float16)xp[p*4+0];
        lo.h[4] = (_Float16)xp[p*4+1];
        lo.h[5] = (_Float16)xp[p*4+2];
        uint4* dst = reinterpret_cast<uint4*>(&sE2[wv][lane * 136]);
        dst[0] = lo.u[0]; dst[1] = lo.u[1]; dst[2] = hi.u[0]; dst[3] = hi.u[1];
    }
    __syncthreads();

    // ---- conv1 GEMM via WMMA: [32,32(pad)]@[32,64] -> sD + b1 ----
    {
        const _Float16* e1p = sE2[wv];
        const v8f zero = {0.f,0.f,0.f,0.f,0.f,0.f,0.f,0.f};
        v16h a0 = load_a_frag(e1p, 0,  0, lane, 136);
        v16h a1 = load_a_frag(e1p, 16, 0, lane, 136);
        float* dp = sD[wv];
        #pragma unroll
        for (int nt = 0; nt < 4; ++nt) {
            v16h bf = load_b_frag(sW1T, nt, 0, lane, 40);
            v8f d0 = __builtin_amdgcn_wmma_f32_16x16x32_f16(false, a0, false, bf, (short)0, zero, false, false);
            v8f d1 = __builtin_amdgcn_wmma_f32_16x16x32_f16(false, a1, false, bf, (short)0, zero, false, false);
            #pragma unroll
            for (int i = 0; i < 8; ++i) {
                int n = nt*16 + (lane & 15);
                int m0 = i + ((lane & 16) ? 8 : 0);
                dp[ m0      *68 + n] = d0[i] + sVec[0*64+n];
                dp[(m0 + 16)*68 + n] = d1[i] + sVec[0*64+n];
            }
        }
    }
    __syncthreads();

    // ---- conv1 LN + SELU + max over k; emit x1 (f32 + f16) ----
    {
        const float* dp = sD[wv];
        float h[64];
        float s = 0.f, s2 = 0.f;
        #pragma unroll
        for (int c = 0; c < 64; ++c) { float a = dp[lane*68+c]; h[c] = a; s += a; s2 += a*a; }
        const float m   = s * (1.f/64.f);
        const float inv = rsqrtf(s2*(1.f/64.f) - m*m + LN_EPS);
        #pragma unroll
        for (int c = 0; c < 64; ++c) {
            float v = selu_f((h[c]-m)*inv*sVec[1*64+c] + sVec[2*64+c]);
            v = fmaxf(v, __shfl_xor(v, 1));
            v = fmaxf(v, __shfl_xor(v, 2));
            h[c] = v;
        }
        const int p = lane >> 2;
        if ((lane & 3) == 0) {
            #pragma unroll
            for (int c = 0; c < 64; ++c) {
                sX1 [wv][p*68+c] = h[c];
                sX1h[wv][p*72+c] = (_Float16)h[c];
            }
        }
    }
    __syncthreads();

    // ---- Gram matrix G = X1 * X1^T via WMMA (16x16x64, rows 8..15 zero) ----
    {
        const _Float16* xh = sX1h[wv];
        v8f g = {0.f,0.f,0.f,0.f,0.f,0.f,0.f,0.f};
        #pragma unroll
        for (int kt = 0; kt < 2; ++kt) {
            const int kk = kt * 32;
            v16h a  = load_a_frag(xh, 0, kk, lane, 72);
            v16h bt = load_b_frag(xh, 0, kk, lane, 72);
            g = __builtin_amdgcn_wmma_f32_16x16x32_f16(false, a, false, bt, (short)0, g, false, false);
        }
        float* dp = sD[wv];
        #pragma unroll
        for (int i = 0; i < 8; ++i) {
            int m = i + ((lane & 16) ? 8 : 0);
            int n = lane & 15;
            dp[m*68+n] = g[i];
        }
    }
    __syncthreads();

    // ---- kNN for conv2 from Gram: d(i,j) = G[i,i] + G[j,j] - 2 G[i,j] ----
    if (lane < 8) {
        const float* gp_ = sD[wv];
        const float gii = gp_[lane*68 + lane];
        float d[8];
        #pragma unroll
        for (int j = 0; j < 8; ++j)
            d[j] = gii + gp_[j*68+j] - 2.f * gp_[lane*68+j];
        unsigned chosen = 0;
        #pragma unroll
        for (int s = 0; s < 4; ++s) {
            float best = 3.4e38f; int bj = 0;
            #pragma unroll
            for (int j = 0; j < 8; ++j)
                if (!((chosen >> j) & 1u) && d[j] < best) { best = d[j]; bj = j; }
            chosen |= 1u << bj;
            idxp[lane*4+s] = bj;
        }
    }
    __syncthreads();

    // ---- build edge features e2 [32,128] f16 ----
    {
        const float* x1p = sX1[wv];
        const int p = lane >> 2, nn = lane & 3;
        const int j = idxp[p*4+nn];
        _Float16* er = &sE2[wv][lane * 136];
        #pragma unroll 8
        for (int c = 0; c < 64; ++c) {
            float a  = x1p[p*68+c];
            float bb = x1p[j*68+c];
            er[c]      = (_Float16)(bb - a);
            er[64 + c] = (_Float16)a;
        }
    }
    __syncthreads();

    // ---- conv2 GEMM via WMMA: [32,128]@[128,64], 2x4 tiles, 4 k-steps ----
    {
        const _Float16* e2p = sE2[wv];
        const v8f zero = {0.f,0.f,0.f,0.f,0.f,0.f,0.f,0.f};
        v8f acc[2][4];
        #pragma unroll
        for (int mt = 0; mt < 2; ++mt)
            #pragma unroll
            for (int nt = 0; nt < 4; ++nt) acc[mt][nt] = zero;

        #pragma unroll
        for (int kt = 0; kt < 4; ++kt) {
            const int kk = kt * 32;
            v16h a0 = load_a_frag(e2p, 0,  kk, lane, 136);
            v16h a1 = load_a_frag(e2p, 16, kk, lane, 136);
            #pragma unroll
            for (int nt = 0; nt < 4; ++nt) {
                v16h bf = load_b_frag(sW2T, nt, kk, lane, 136);
                acc[0][nt] = __builtin_amdgcn_wmma_f32_16x16x32_f16(
                    false, a0, false, bf, (short)0, acc[0][nt], false, false);
                acc[1][nt] = __builtin_amdgcn_wmma_f32_16x16x32_f16(
                    false, a1, false, bf, (short)0, acc[1][nt], false, false);
            }
        }
        float* dp = sD[wv];
        #pragma unroll
        for (int mt = 0; mt < 2; ++mt)
            #pragma unroll
            for (int nt = 0; nt < 4; ++nt)
                #pragma unroll
                for (int i = 0; i < 8; ++i) {
                    int m = mt*16 + i + ((lane & 16) ? 8 : 0);
                    int n = nt*16 + (lane & 15);
                    dp[m*68+n] = acc[mt][nt][i] + sVec[3*64+n];
                }
    }
    __syncthreads();

    // ---- conv2 LN + SELU + max over k ----
    {
        const float* dp = sD[wv];
        float h[64];
        float s = 0.f, s2 = 0.f;
        #pragma unroll
        for (int c = 0; c < 64; ++c) { float a = dp[lane*68+c]; h[c] = a; s += a; s2 += a*a; }
        const float m   = s * (1.f/64.f);
        const float inv = rsqrtf(s2*(1.f/64.f) - m*m + LN_EPS);
        #pragma unroll
        for (int c = 0; c < 64; ++c) {
            float v = selu_f((h[c]-m)*inv*sVec[4*64+c] + sVec[5*64+c]);
            v = fmaxf(v, __shfl_xor(v, 1));
            v = fmaxf(v, __shfl_xor(v, 2));
            h[c] = v;
        }
        const int p = lane >> 2;
        if ((lane & 3) == 0) {
            #pragma unroll
            for (int c = 0; c < 64; ++c) sX2[wv][p*68+c] = h[c];
        }
    }
    __syncthreads();

    // ---- pool concat(x1,x2) over points -> f16 feat row for this box ----
    {
        #pragma unroll
        for (int t = 0; t < 4; ++t) {
            int c = lane*4 + t;
            float m = -3.4e38f;
            #pragma unroll
            for (int p = 0; p < 8; ++p) {
                float v = (c < 64) ? sX1[wv][p*68+c] : sX2[wv][p*68+(c-64)];
                m = fmaxf(m, v);
            }
            sFEATH[wv*136 + c] = (_Float16)m;
        }
    }
    __syncthreads();

    // ---- block projection via WMMA (wave 0): [16(pad),128]@[128,64] ----
    if (wv == 0) {
        const v8f zero = {0.f,0.f,0.f,0.f,0.f,0.f,0.f,0.f};
        v8f pacc[4];
        #pragma unroll
        for (int nt = 0; nt < 4; ++nt) pacc[nt] = zero;
        #pragma unroll
        for (int kt = 0; kt < 4; ++kt) {
            const int kk = kt * 32;
            v16h a = load_a_frag(sFEATH, 0, kk, lane, 136);
            #pragma unroll
            for (int nt = 0; nt < 4; ++nt) {
                v16h bf = load_b_frag(sWpT, nt, kk, lane, 136);
                pacc[nt] = __builtin_amdgcn_wmma_f32_16x16x32_f16(
                    false, a, false, bf, (short)0, pacc[nt], false, false);
            }
        }
        #pragma unroll
        for (int nt = 0; nt < 4; ++nt)
            #pragma unroll
            for (int i = 0; i < 8; ++i) {
                int m = i + ((lane & 16) ? 8 : 0);
                int n = nt*16 + (lane & 15);
                sPD[m*68+n] = pacc[nt][i] + sVec[6*64+n];
            }
    }
    __syncthreads();

    // ---- final LN (wave-wide shuffle reduce) + SELU ----
    {
        const int c0 = lane*2, c1 = c0 + 1;
        float a0 = sPD[wv*68+c0], a1 = sPD[wv*68+c1];
        float s = a0 + a1, s2 = a0*a0 + a1*a1;
        #pragma unroll
        for (int off = 16; off >= 1; off >>= 1) {
            s  += __shfl_xor(s,  off);
            s2 += __shfl_xor(s2, off);
        }
        const float m   = s * (1.f/64.f);
        const float inv = rsqrtf(s2*(1.f/64.f) - m*m + LN_EPS);
        out[box*64+c0] = selu_f((a0-m)*inv*sVec[7*64+c0] + sVec[8*64+c0]);
        out[box*64+c1] = selu_f((a1-m)*inv*sVec[7*64+c1] + sVec[8*64+c1]);
    }
}

extern "C" void kernel_launch(void* const* d_in, const int* in_sizes, int n_in,
                              void* d_out, int out_size, void* d_ws, size_t ws_size,
                              hipStream_t stream) {
    (void)in_sizes; (void)n_in; (void)d_ws; (void)ws_size; (void)out_size;
    const float* x   = (const float*)d_in[0];
    const float* W1  = (const float*)d_in[1];
    const float* b1  = (const float*)d_in[2];
    const float* g1  = (const float*)d_in[3];
    const float* be1 = (const float*)d_in[4];
    const float* W2  = (const float*)d_in[5];
    const float* b2  = (const float*)d_in[6];
    const float* g2  = (const float*)d_in[7];
    const float* be2 = (const float*)d_in[8];
    const float* Wp  = (const float*)d_in[9];
    const float* bp  = (const float*)d_in[10];
    const float* gp  = (const float*)d_in[11];
    const float* bep = (const float*)d_in[12];
    float* out = (float*)d_out;

    dim3 grid(B_TOTAL / NWAVE), block(32 * NWAVE);
    hipLaunchKernelGGL(bbox_encoder_kernel, grid, block, 0, stream,
                       x, W1, b1, g1, be1, W2, b2, g2, be2, Wp, bp, gp, bep, out);
}
